// PoseNetXOX_82772609729158
// MI455X (gfx1250) — compile-verified, matrix operand order checked by reading, not run on hardware
//
#include <hip/hip_runtime.h>
#include <hip/hip_bf16.h>

typedef __attribute__((ext_vector_type(16))) _Float16 v16h;
typedef __attribute__((ext_vector_type(8)))  float    v8f;

#define NNODES 8
#define NEDGES 64
#define HWDIM 256
#define KDIM 294      // 6*7*7
#define NCHUNK 10     // K padded to 320 = 10 * 32
#define COUT 64
#define FEAT 1024

// workspace layout (bytes)
#define WS_BPACK   0         // 10*4*32*16 halves = 40960 B
#define WS_SCALE   40960     // 64 f32
#define WS_SHIFT   41216     // 64 f32
#define WS_POOLED  41472     // 64*64 f32 = 16384 B
#define WS_EF      57856     // 64*1024 f32 = 262144 B  -> total 320000 B

// K pattern for 16-bit WMMA A/B fragments (ISA 7.12.2):
// slot s = vgpr*2 + half ; khalf = lane>>4
//   k = chunk*32 + basepat(s) + khalf*8,  basepat(s) = (s<8) ? s : s+8
__device__ __forceinline__ int kpat(int khalf, int s) {
    return (s < 8 ? s : s + 8) + khalf * 8;
}

// LDS offset (in halves) of im2col element k for conv row 0; add row*524 for rows 1,2.
// Input staged as [c:6][s':11][col:262], s' = 2*row + ky, col = 2*ox + kx.
__device__ __forceinline__ int ldsoff(int k) {
    if (k >= KDIM) return -1;
    int c = k / 49, r = k - c * 49, ky = r / 7, kx = r - ky * 7;
    return c * 2882 + ky * 262 + kx;
}

__global__ void prep_kernel(const float* __restrict__ w, const float* __restrict__ g,
                            const float* __restrict__ be, const float* __restrict__ mu,
                            const float* __restrict__ var,
                            _Float16* __restrict__ Bpack, float* __restrict__ scale,
                            float* __restrict__ shift, float* __restrict__ pooled) {
    int gtid = blockIdx.x * blockDim.x + threadIdx.x;
    if (gtid < NCHUNK * 4 * 32 * 16) {
        int slot  = gtid & 15;
        int lane  = (gtid >> 4) & 31;
        int nt    = (gtid >> 9) & 3;
        int chunk = gtid >> 11;
        int n = nt * 16 + (lane & 15);
        int k = chunk * 32 + kpat(lane >> 4, slot);
        _Float16 val = (_Float16)0.f;
        if (k < KDIM) val = (_Float16)w[n * KDIM + k];   // B[k][n] = conv1_w[n][k]
        Bpack[gtid] = val;
    }
    if (gtid < COUT) {
        float inv = rsqrtf(var[gtid] + 1e-5f);
        float sc = g[gtid] * inv;
        scale[gtid] = sc;
        shift[gtid] = be[gtid] - mu[gtid] * sc;
    }
    if (gtid < NEDGES * COUT) pooled[gtid] = 0.f;
}

// One block per (edge, pooled-row py). Computes conv rows oy = 2py-1..2py+1 via
// implicit-GEMM WMMA, fuses BN+ReLU+maxpool(3x3,s2,p1)+avg partial-sum.
__launch_bounds__(256)
__global__ void conv_pool_kernel(const float* __restrict__ x, const int* __restrict__ ei,
                                 const _Float16* __restrict__ Bpack,
                                 const float* __restrict__ scale,
                                 const float* __restrict__ shift,
                                 float* __restrict__ pooled) {
    // Reused buffer: phase 1 = staged input (6ch x 11 rows x 262 cols = 17292 halves)
    //                phase 2 = conv output  (3 rows x 128 ox x 64 ch = 24576 halves)
    __shared__ _Float16 smem[3 * 128 * 64];
    __shared__ float red[256];

    const int py  = blockIdx.x;
    const int e   = blockIdx.y;
    const int tid = threadIdx.x;

    int i0 = ei[e], i1 = ei[NEDGES + e];
    int nmin = i0 < i1 ? i0 : i1;
    int nmax = i0 < i1 ? i1 : i0;

    // ---- stage input rows iy = 4py-5 .. 4py+5, cols ix = -3..258, zero-padded, f32->f16
    for (int i = tid; i < 6 * 11 * 262; i += 256) {
        int c = i / 2882;  int rem = i - c * 2882;      // 2882 = 11*262
        int s = rem / 262; int col = rem - s * 262;
        int iy = 4 * py - 5 + s;
        int ix = col - 3;
        float v = 0.f;
        if (iy >= 0 && iy < HWDIM && ix >= 0 && ix < HWDIM) {
            int node = (c < 3) ? nmin : nmax;
            int ch   = (c < 3) ? c : c - 3;
            v = x[node * (3 * HWDIM * HWDIM) + ch * (HWDIM * HWDIM) + iy * HWDIM + ix];
        }
        smem[i] = (_Float16)v;
    }
    __syncthreads();

    const int lane  = tid & 31;
    const int wave  = tid >> 5;       // 8 waves, each owns ox tile [wave*16, wave*16+16)
    const int khalf = lane >> 4;
    const int nlo   = lane & 15;
    const int ox    = wave * 16 + nlo;   // A-matrix row M = nlo -> this output x
    const int ox2   = 2 * ox;

    v8f acc[3][4] = {};
    union AF { v16h v; _Float16 h[16]; };

    #pragma unroll
    for (int chunk = 0; chunk < NCHUNK; ++chunk) {
        AF A[3];
        #pragma unroll
        for (int s = 0; s < 16; ++s) {
            int k0 = chunk * 32 + (s < 8 ? s : s + 8);  // khalf = 0 variant
            int o0 = ldsoff(k0);                        // compile-time after unroll
            int o1 = ldsoff(k0 + 8);                    // khalf = 1 variant
            if (o0 >= 0 && o1 >= 0) {                   // chunks 0..8: always here
                int off = ox2 + (khalf ? o1 : o0);      // cndmask of two constants
                #pragma unroll
                for (int row = 0; row < 3; ++row)
                    A[row].h[s] = smem[off + row * 524];  // ds_load imm offsets
            } else if (o0 >= 0) {                       // chunk 9 tail, khalf=0 only
                int off = ox2 + o0;
                #pragma unroll
                for (int row = 0; row < 3; ++row)
                    A[row].h[s] = khalf ? (_Float16)0.f : smem[off + row * 524];
            } else {                                    // fully padded slot
                #pragma unroll
                for (int row = 0; row < 3; ++row)
                    A[row].h[s] = (_Float16)0.f;
            }
        }
        #pragma unroll
        for (int nt = 0; nt < 4; ++nt) {
            v16h B = *(const v16h*)(Bpack + ((chunk * 4 + nt) * 32 + lane) * 16);
            #pragma unroll
            for (int row = 0; row < 3; ++row) {
                acc[row][nt] = __builtin_amdgcn_wmma_f32_16x16x32_f16(
                    false, A[row].v, false, B, (short)0, acc[row][nt], false, false);
            }
        }
    }
    __syncthreads();   // all waves done reading staged input; smem is reused below

    // ---- BN + ReLU, scatter C fragments into smem[row][ox][c] (f16, channels-last)
    #pragma unroll
    for (int nt = 0; nt < 4; ++nt) {
        int c = nt * 16 + nlo;            // C layout: N = lane&15
        float sc = scale[c], sh = shift[c];
        #pragma unroll
        for (int row = 0; row < 3; ++row) {
            #pragma unroll
            for (int j = 0; j < 8; ++j) { // C layout: M = j + 8*(lane>=16)
                int oxw = wave * 16 + j + 8 * khalf;
                float v = fmaxf(acc[row][nt][j] * sc + sh, 0.f);
                smem[(row * 128 + oxw) * 64 + c] = (_Float16)v;
            }
        }
    }
    __syncthreads();

    // ---- maxpool 3x3 s2 p1 over the 3 conv rows, partial sum over 64 px for avg-pool
    // post-ReLU values are >= 0 so 0 stands in for the -inf padding identity.
    int c = tid & 63, grp = tid >> 6;
    int rlo = (py == 0) ? 1 : 0;          // oy = -1 row excluded at py==0
    float sum = 0.f;
    for (int px = grp * 16; px < grp * 16 + 16; ++px) {
        int xlo = (px == 0) ? 0 : 2 * px - 1;
        int xhi = 2 * px + 1;             // <= 127 always
        float m = 0.f;
        for (int row = rlo; row < 3; ++row)
            for (int oxp = xlo; oxp <= xhi; ++oxp)
                m = fmaxf(m, (float)smem[(row * 128 + oxp) * 64 + c]);
        sum += m;
    }
    red[tid] = sum;
    __syncthreads();
    if (tid < 64)
        atomicAdd(&pooled[e * 64 + tid],
                  red[tid] + red[64 + tid] + red[128 + tid] + red[192 + tid]);
}

__launch_bounds__(256)
__global__ void fc1_kernel(const float* __restrict__ pooled, const float* __restrict__ fw,
                           const float* __restrict__ fb, float* __restrict__ ef) {
    __shared__ float p[64];
    int e = blockIdx.x, tid = threadIdx.x;
    if (tid < 64) p[tid] = pooled[e * 64 + tid] * (1.f / 4096.f);  // avg over 64*64
    __syncthreads();
    for (int f = tid; f < FEAT; f += 256) {
        float s = fb[f];
        #pragma unroll 8
        for (int c = 0; c < 64; ++c) s += p[c] * fw[f * 64 + c];
        ef[e * FEAT + f] = fmaxf(s, 0.f);
    }
}

__global__ void heads_kernel(const float* __restrict__ ef,
                             const float* __restrict__ wxyz, const float* __restrict__ bxyz,
                             const float* __restrict__ wpqr, const float* __restrict__ bpqr,
                             float* __restrict__ out) {
    int e = blockIdx.x, tid = threadIdx.x;
    if (tid < 6) {
        const float* wv = (tid < 3) ? (wxyz + tid * FEAT) : (wpqr + (tid - 3) * FEAT);
        float s = (tid < 3) ? bxyz[tid] : bpqr[tid - 3];
        const float* ev = ef + e * FEAT;
        for (int f = 0; f < FEAT; ++f) s += ev[f] * wv[f];
        out[NNODES * 6 + e * 6 + tid] = s;     // full [64,6]
        if (e < NNODES) out[e * 6 + tid] = s;  // full[:8]
    }
}

extern "C" void kernel_launch(void* const* d_in, const int* in_sizes, int n_in,
                              void* d_out, int out_size, void* d_ws, size_t ws_size,
                              hipStream_t stream) {
    const float* x      = (const float*)d_in[0];
    const int*   ei     = (const int*)  d_in[1];
    const float* conv1w = (const float*)d_in[2];
    const float* bng    = (const float*)d_in[3];
    const float* bnb    = (const float*)d_in[4];
    const float* bnm    = (const float*)d_in[5];
    const float* bnv    = (const float*)d_in[6];
    const float* fcw    = (const float*)d_in[7];
    const float* fcb    = (const float*)d_in[8];
    const float* wxyz   = (const float*)d_in[9];
    const float* bxyz   = (const float*)d_in[10];
    const float* wpqr   = (const float*)d_in[11];
    const float* bpqr   = (const float*)d_in[12];
    float* out = (float*)d_out;

    char* ws = (char*)d_ws;
    _Float16* Bpack = (_Float16*)(ws + WS_BPACK);
    float* scale  = (float*)(ws + WS_SCALE);
    float* shift  = (float*)(ws + WS_SHIFT);
    float* pooled = (float*)(ws + WS_POOLED);
    float* ef     = (float*)(ws + WS_EF);

    prep_kernel<<<80, 256, 0, stream>>>(conv1w, bng, bnb, bnm, bnv,
                                        Bpack, scale, shift, pooled);
    conv_pool_kernel<<<dim3(64, 64), 256, 0, stream>>>(x, ei, Bpack, scale, shift, pooled);
    fc1_kernel<<<64, 256, 0, stream>>>(pooled, fcw, fcb, ef);
    heads_kernel<<<64, 32, 0, stream>>>(ef, wxyz, bxyz, wpqr, bpqr, out);
}
